// LSTModel_21938692948472
// MI455X (gfx1250) — compile-verified, matrix operand order checked by reading, not run on hardware
//
#include <hip/hip_runtime.h>
#include <math.h>

typedef __bf16 v16bf __attribute__((ext_vector_type(16)));
typedef float  v8f   __attribute__((ext_vector_type(8)));

#define VOCAB 50000
#define EMBED 128
#define SEQ   200
#define HID   30
#define HID_P 32
#define BATCH 4096

#define WMMA_BF16(A, B, C) \
  __builtin_amdgcn_wmma_f32_16x16x32_bf16(false, (A), false, (B), (short)0, (C), false, false)

// ---------------------------------------------------------------------------
// Hardware tanh (CDNA5 TRANS op). Builtin when available (compiler handles the
// TRANS hazard); otherwise inline asm with a v_nop to satisfy the 1-op gap.
// ---------------------------------------------------------------------------
__device__ __forceinline__ float fast_tanh(float x) {
#if __has_builtin(__builtin_amdgcn_tanhf)
  return __builtin_amdgcn_tanhf(x);
#elif __has_builtin(__builtin_amdgcn_tanh_f32)
  return __builtin_amdgcn_tanh_f32(x);
#else
  float y;
  asm("v_tanh_f32 %0, %1\n\tv_nop" : "=v"(y) : "v"(x));
  return y;
#endif
}

// ---------------------------------------------------------------------------
// Branchless B-tile load: 32x16 (KxN) tile of a row-major [kmax x ncols] f32
// matrix into bf16 B-layout, zero-padding k >= kmax and n >= nmax via masks
// (clamped index + 0/1 multiply -> v_cndmask, no EXEC divergence).
// ---------------------------------------------------------------------------
__device__ __forceinline__ v16bf load_b_tile(const float* __restrict__ W, int ncols,
                                             int k0, int n, int kmax, int nmax,
                                             int koff) {
  const int   nc = (n < nmax) ? n : 0;
  const float mn = (n < nmax) ? 1.f : 0.f;
  v16bf b;
#pragma unroll
  for (int i = 0; i < 8; ++i) {
    const int ka = k0 + koff + i;
    const int kb = k0 + koff + 16 + i;
    const int kac = (ka < kmax) ? ka : 0;
    const int kbc = (kb < kmax) ? kb : 0;
    const float ma = (ka < kmax) ? mn : 0.f;
    const float mb = (kb < kmax) ? mn : 0.f;
    b[i]     = (__bf16)(W[kac * ncols + nc] * ma);
    b[8 + i] = (__bf16)(W[kbc * ncols + nc] * mb);
  }
  return b;
}

// A-tile (16x32, bf16) from an LDS row in 16-bit A-layout: two 16B chunks.
__device__ __forceinline__ v16bf load_a_lds(const __bf16* rowPtr, int k0, int koff) {
  union { v16bf v; uint4 q[2]; } au;
  const uint4* p = reinterpret_cast<const uint4*>(rowPtr + k0);
  au.q[0] = p[koff >> 3];          // K = k0+koff .. +7
  au.q[1] = p[(koff + 16) >> 3];   // K = k0+koff+16 .. +23
  return au.v;
}

// ---------------------------------------------------------------------------
// Kernel A: P[v][c] = sum_e emb[v][e] * Wx[e][c] + b_rnn[c]  (c < 30, else 0)
// One wave32 per 16 vocab rows; bf16 WMMA, K=128 in 4 chunks, N 30->32 padded.
// ---------------------------------------------------------------------------
__global__ __launch_bounds__(32) void build_table(
    const float* __restrict__ emb,
    const float* __restrict__ Wx,
    const float* __restrict__ b_rnn,
    float* __restrict__ P) {
  const int l      = threadIdx.x;
  const int lane16 = l & 15;
  const int hi     = l >> 4;
  const int koff   = hi ? 8 : 0;
  const int vbase  = blockIdx.x * 16;
  const int row    = vbase + lane16;

  const int n0 = lane16;
  const int n1 = 16 + lane16;

  v8f acc0 = {}; v8f acc1 = {};
  for (int kc = 0; kc < 4; ++kc) {
    const int k0 = kc * 32;
    v16bf a;
    const float* ep = emb + (long)row * EMBED + k0 + koff;
#pragma unroll
    for (int i = 0; i < 8; ++i) a[i]     = (__bf16)ep[i];
#pragma unroll
    for (int i = 0; i < 8; ++i) a[8 + i] = (__bf16)ep[16 + i];
    const v16bf b0 = load_b_tile(Wx, HID, k0, n0, EMBED, HID, koff);
    const v16bf b1 = load_b_tile(Wx, HID, k0, n1, EMBED, HID, koff);
    acc0 = WMMA_BF16(a, b0, acc0);
    acc1 = WMMA_BF16(a, b1, acc1);
  }
  // Branchless bias: unconditional load from clamped index, 0/1 mask multiply
  const float bias0 = b_rnn[n0];                               // n0 < 16 valid
  const float bias1 = b_rnn[(n1 < HID) ? n1 : 0] * ((n1 < HID) ? 1.f : 0.f);
#pragma unroll
  for (int r = 0; r < 8; ++r) {
    const int vr = vbase + r + 8 * hi;
    P[(long)vr * HID_P + n0] = acc0[r] + bias0;
    P[(long)vr * HID_P + n1] = acc1[r] + bias1;       // padded cols: 0 + 0
  }
}

// ---------------------------------------------------------------------------
// Kernel B: 200-step tanh recurrence + WMMA MLP head. One wave32 / 16 rows.
// ---------------------------------------------------------------------------
__global__ __launch_bounds__(32) void rnn_head(
    const int*   __restrict__ tokens,
    const float* __restrict__ P,
    const float* __restrict__ Wh,
    const float* __restrict__ W1, const float* __restrict__ B1,
    const float* __restrict__ W2, const float* __restrict__ B2,
    const float* __restrict__ W3, const float* __restrict__ B3,
    const float* __restrict__ Wo, const float* __restrict__ Bo,
    float* __restrict__ out) {
  __shared__ __attribute__((aligned(16))) __bf16 hb[16][HID_P];   // h staging
  __shared__ __attribute__((aligned(16))) __bf16 d1s[16][128];
  __shared__ __attribute__((aligned(16))) __bf16 d2s[16][64];
  __shared__ __attribute__((aligned(16))) __bf16 d3s[16][32];

  const int l      = threadIdx.x;
  const int lane16 = l & 15;
  const int hi     = l >> 4;
  const int koff   = hi ? 8 : 0;
  const int base   = blockIdx.x * 16;

  // Wh in B-layout, zero-padded to 32x32 (branchless)
  const v16bf wb0 = load_b_tile(Wh, HID, 0, lane16,      HID, HID, koff);
  const v16bf wb1 = load_b_tile(Wh, HID, 0, 16 + lane16, HID, HID, koff);

  // h0 = 0
#pragma unroll
  for (int r = 0; r < 8; ++r) {
    const int row = r + 8 * hi;
    hb[row][lane16]      = (__bf16)0.f;
    hb[row][16 + lane16] = (__bf16)0.f;
  }
  __syncthreads();

  const long tokRowBase = (long)(base + lane16) * SEQ;

  // Gather xp_t = P[tok] rows straight into WMMA C layout (row via ds_bpermute)
  auto gather_xp = [&](int tokLane, v8f& c0, v8f& c1) {
#pragma unroll
    for (int r = 0; r < 8; ++r) {
      const int tok = __shfl(tokLane, r + 8 * hi, 32);
      const float* prow = P + (long)tok * HID_P;
      c0[r] = prow[lane16];
      c1[r] = prow[16 + lane16];
    }
  };

  v8f c0, c1;
  gather_xp(tokens[tokRowBase], c0, c1);

  for (int t = 0; t < SEQ; ++t) {
    // A = h_{t-1} (bf16) from LDS
    const v16bf a = load_a_lds(&hb[lane16][0], 0, koff);
    const v8f a0 = WMMA_BF16(a, wb0, c0);
    const v8f a1 = WMMA_BF16(a, wb1, c1);

    // Software-pipeline: fetch next step's xp while WMMA/tanh are in flight
    if (t + 1 < SEQ) {
      gather_xp(tokens[tokRowBase + t + 1], c0, c1);
    }

    __syncthreads();  // all A-reads for this step done before overwrite
#pragma unroll
    for (int r = 0; r < 8; ++r) {
      const int row = r + 8 * hi;
      hb[row][lane16]      = (__bf16)fast_tanh(a0[r]);
      hb[row][16 + lane16] = (__bf16)fast_tanh(a1[r]);
    }
    __syncthreads();  // new h visible before next step's A-reads
  }

  // ---------------- WMMA MLP head (h_last lives in hb) ----------------
  // L1: [16x30] @ [30x128] -> relu -> d1s
  {
    const v16bf ah = load_a_lds(&hb[lane16][0], 0, koff);
#pragma unroll
    for (int nt = 0; nt < 8; ++nt) {
      const int n = nt * 16 + lane16;
      const float bias = B1[n];
      v8f acc;
#pragma unroll
      for (int r = 0; r < 8; ++r) acc[r] = bias;
      const v16bf b = load_b_tile(W1, 128, 0, n, HID, 128, koff);
      acc = WMMA_BF16(ah, b, acc);
#pragma unroll
      for (int r = 0; r < 8; ++r)
        d1s[r + 8 * hi][n] = (__bf16)(acc[r] > 0.f ? acc[r] : 0.f);
    }
  }
  __syncthreads();

  // L2: [16x128] @ [128x64] -> relu -> d2s
#pragma unroll
  for (int nt = 0; nt < 4; ++nt) {
    const int n = nt * 16 + lane16;
    const float bias = B2[n];
    v8f acc;
#pragma unroll
    for (int r = 0; r < 8; ++r) acc[r] = bias;
#pragma unroll
    for (int kc = 0; kc < 4; ++kc) {
      const v16bf a = load_a_lds(&d1s[lane16][0], kc * 32, koff);
      const v16bf b = load_b_tile(W2, 64, kc * 32, n, 128, 64, koff);
      acc = WMMA_BF16(a, b, acc);
    }
#pragma unroll
    for (int r = 0; r < 8; ++r)
      d2s[r + 8 * hi][n] = (__bf16)(acc[r] > 0.f ? acc[r] : 0.f);
  }
  __syncthreads();

  // L3: [16x64] @ [64x32] -> relu -> d3s
#pragma unroll
  for (int nt = 0; nt < 2; ++nt) {
    const int n = nt * 16 + lane16;
    const float bias = B3[n];
    v8f acc;
#pragma unroll
    for (int r = 0; r < 8; ++r) acc[r] = bias;
#pragma unroll
    for (int kc = 0; kc < 2; ++kc) {
      const v16bf a = load_a_lds(&d2s[lane16][0], kc * 32, koff);
      const v16bf b = load_b_tile(W3, 32, kc * 32, n, 64, 32, koff);
      acc = WMMA_BF16(a, b, acc);
    }
#pragma unroll
    for (int r = 0; r < 8; ++r)
      d3s[r + 8 * hi][n] = (__bf16)(acc[r] > 0.f ? acc[r] : 0.f);
  }
  __syncthreads();

  // Out: [16x32] @ [32x1] -> sigmoid (only column 0 of the tile is real)
  {
    const float bz   = Bo[0];                      // uniform scalar load
    const float bias = (lane16 == 0) ? bz : 0.f;   // branchless select
    v8f acc;
#pragma unroll
    for (int r = 0; r < 8; ++r) acc[r] = bias;
    const v16bf a = load_a_lds(&d3s[lane16][0], 0, koff);
    const v16bf b = load_b_tile(Wo, 1, 0, lane16, 32, 1, koff);
    acc = WMMA_BF16(a, b, acc);
    if (lane16 == 0) {
#pragma unroll
      for (int r = 0; r < 8; ++r)
        out[base + r + 8 * hi] = 1.f / (1.f + __expf(-acc[r]));
    }
  }
}

// ---------------------------------------------------------------------------
extern "C" void kernel_launch(void* const* d_in, const int* in_sizes, int n_in,
                              void* d_out, int out_size, void* d_ws, size_t ws_size,
                              hipStream_t stream) {
  (void)in_sizes; (void)n_in; (void)out_size; (void)ws_size;
  const int*   tokens = (const int*)  d_in[0];
  const float* emb    = (const float*)d_in[1];
  const float* Wx     = (const float*)d_in[2];
  const float* Wh     = (const float*)d_in[3];
  const float* b_rnn  = (const float*)d_in[4];
  const float* W1     = (const float*)d_in[5];
  const float* b1     = (const float*)d_in[6];
  const float* W2     = (const float*)d_in[7];
  const float* b2     = (const float*)d_in[8];
  const float* W3     = (const float*)d_in[9];
  const float* b3     = (const float*)d_in[10];
  const float* Wo     = (const float*)d_in[11];
  const float* bo     = (const float*)d_in[12];

  float* P = (float*)d_ws;  // [VOCAB][32] f32 = 6.4 MB, L2-resident gather table

  build_table<<<VOCAB / 16, 32, 0, stream>>>(emb, Wx, b_rnn, P);
  rnn_head<<<BATCH / 16, 32, 0, stream>>>(tokens, P, Wh,
                                          W1, b1, W2, b2, W3, b3, Wo, bo,
                                          (float*)d_out);
}
